// EdgeP_43748536877307
// MI455X (gfx1250) — matrix operand order repaired and checked by reading, not compiled, not run on hardware
//
#include <hip/hip_runtime.h>
#include <hip/hip_bf16.h>

#define N_NODES 50000
#define DIM     128
#define NEDGE   800000
#define NLAYER  3

#define XPITCH  (2 * DIM + 8)   // padded xcat row (halfs): +16B -> 4-bank shift/row
#define HPITCH  (DIM + 8)       // padded hidden row (halfs)

typedef __attribute__((ext_vector_type(16))) _Float16 v16h;
typedef __attribute__((ext_vector_type(8)))  _Float16 v8h;
typedef __attribute__((ext_vector_type(8)))  float    v8f;

// ---- monotone order-preserving encoding of f32 into u32 (for atomic max) ----
__device__ __forceinline__ unsigned fkey(float f) {
    unsigned u = __float_as_uint(f);
    return (u & 0x80000000u) ? ~u : (u | 0x80000000u);
}
__device__ __forceinline__ float fdec(unsigned k) {
    unsigned u = (k & 0x80000000u) ? (k & 0x7FFFFFFFu) : ~k;
    return __uint_as_float(u);
}
#define NEG_INF_KEY 0x007FFFFFu   // fkey(-inf)

// ---- pack W1/W2 (f32, input-major [K,N]) into WMMA B-fragment f16 layout ----
// pw[(kt*128 + n)*32 + k32] = W[kt*32 + k32][n]; lane (n + 16*g) reads 16
// contiguous halfs at +g*16 => B rows K=g*16..g*16+15 of column n.
__global__ void pack_weights_kernel(const float* __restrict__ W1,
                                    const float* __restrict__ W2,
                                    _Float16* __restrict__ pw1,
                                    _Float16* __restrict__ pw2) {
    const int nW1 = NLAYER * 2 * DIM * DIM;   // 98304
    const int nW2 = NLAYER * DIM * DIM;       // 49152
    for (int i = blockIdx.x * blockDim.x + threadIdx.x; i < nW1 + nW2;
         i += gridDim.x * blockDim.x) {
        if (i < nW1) {
            int l = i / (2 * DIM * DIM);
            int r = i % (2 * DIM * DIM);
            int kk = r / DIM, n = r % DIM;
            int kt = kk >> 5, k32 = kk & 31;
            pw1[(((size_t)(l * 8 + kt) * DIM) + n) * 32 + k32] = (_Float16)W1[i];
        } else {
            int j = i - nW1;
            int l = j / (DIM * DIM);
            int r = j % (DIM * DIM);
            int kk = r / DIM, n = r % DIM;
            int kt = kk >> 5, k32 = kk & 31;
            pw2[(((size_t)(l * 4 + kt) * DIM) + n) * 32 + k32] = (_Float16)W2[j];
        }
    }
}

__global__ void init_keys_kernel(unsigned* __restrict__ keys) {
    for (int i = blockIdx.x * blockDim.x + threadIdx.x; i < N_NODES * DIM;
         i += gridDim.x * blockDim.x)
        keys[i] = NEG_INF_KEY;
}

__global__ void finalize_kernel(const unsigned* __restrict__ keys,
                                float* __restrict__ out, int do_relu) {
    const float inf = __builtin_inff();
    for (int i = blockIdx.x * blockDim.x + threadIdx.x; i < N_NODES * DIM;
         i += gridDim.x * blockDim.x) {
        float v = fdec(keys[i]);
        if (!(v > -inf && v < inf)) v = 0.0f;       // -inf (empty) / nan -> 0
        if (do_relu) v = v > 0.0f ? v : 0.0f;
        out[i] = v;
    }
}

// ---- per-edge 2-layer MLP via v_wmma_f32_16x16x32_f16 + atomic-max scatter ----
// 2 waves / block; each wave owns 32 edges (two 16-edge M-tiles) so every
// B-fragment load feeds two WMMAs (halves L2 weight traffic).
__global__ __launch_bounds__(64) void edge_mlp_kernel(
    const float* __restrict__ h_in,
    const long long* __restrict__ eidx,        // [2, E] int64: row0=src, row1=dst
    const _Float16* __restrict__ pw1,          // this layer, packed [8][128][32]
    const float* __restrict__ b1,              // [128]
    const _Float16* __restrict__ pw2,          // this layer, packed [4][128][32]
    const float* __restrict__ b2,              // [128]
    unsigned* __restrict__ keys)               // [N,128] encoded running max
{
    __shared__ __align__(32) _Float16 s_xcat[2][32][XPITCH]; // ~33 KB
    __shared__ __align__(32) _Float16 s_hid[2][32][HPITCH];  // ~17 KB
    __shared__ int s_dst[2][32];

    const int w     = threadIdx.x >> 5;
    const int lane  = threadIdx.x & 31;
    const int ebase = (blockIdx.x * 2 + w) * 32;

    // warm the packed weights into cache (global_prefetch_b8)
    __builtin_prefetch(pw1 + (size_t)lane * 512, 0, 1);
    __builtin_prefetch(pw2 + (size_t)lane * 512, 0, 1);

    // Gather: lane l owns edge (ebase+l): dst row -> cols 0..127, src -> 128..255
    {
        const int e = ebase + lane;
        const long long nd = eidx[NEDGE + e];
        const long long ns = eidx[e];
        s_dst[w][lane] = (int)nd;
        const float4* rd = (const float4*)(h_in + (size_t)nd * DIM);
        const float4* rs = (const float4*)(h_in + (size_t)ns * DIM);
        _Float16* dp = &s_xcat[w][lane][0];
        #pragma unroll
        for (int k4 = 0; k4 < DIM / 4; ++k4) {
            float4 v = rd[k4];
            dp[k4 * 4 + 0] = (_Float16)v.x;
            dp[k4 * 4 + 1] = (_Float16)v.y;
            dp[k4 * 4 + 2] = (_Float16)v.z;
            dp[k4 * 4 + 3] = (_Float16)v.w;
        }
        #pragma unroll
        for (int k4 = 0; k4 < DIM / 4; ++k4) {
            float4 v = rs[k4];
            dp[DIM + k4 * 4 + 0] = (_Float16)v.x;
            dp[DIM + k4 * 4 + 1] = (_Float16)v.y;
            dp[DIM + k4 * 4 + 2] = (_Float16)v.z;
            dp[DIM + k4 * 4 + 3] = (_Float16)v.w;
        }
    }
    __syncthreads();

    const int g  = lane >> 4;   // lane group (selects K half of A / K half of B)
    const int nl = lane & 15;   // A-row-in-tile M / B,C,D column-in-tile

    // A fragments, both M-tiles, GEMM1. contiguous K order used here (matches packed B):
    // elems 0..7 : K = kb + g*16 + i ; elems 8..15 : K = kb + g*16 + 8 + i.
    v16h afr[2][8];
    #pragma unroll
    for (int mt = 0; mt < 2; ++mt) {
        #pragma unroll
        for (int kt = 0; kt < 8; ++kt) {
            const int kb = kt * 32;
            v8h lo = *(const v8h*)&s_xcat[w][mt * 16 + nl][kb + g * 16];
            v8h hi = *(const v8h*)&s_xcat[w][mt * 16 + nl][kb + g * 16 + 8];
            v16h a;
            #pragma unroll
            for (int i = 0; i < 8; ++i) { a[i] = lo[i]; a[8 + i] = hi[i]; }
            afr[mt][kt] = a;
        }
    }

    // GEMM1: hidden[32x128] = relu(xcat[32x256] @ W1 + b1)
    for (int nt = 0; nt < 8; ++nt) {
        const int col = nt * 16 + nl;
        const float bv = b1[col];
        v8f c0, c1;
        #pragma unroll
        for (int r = 0; r < 8; ++r) { c0[r] = bv; c1[r] = bv; }
        #pragma unroll
        for (int kt = 0; kt < 8; ++kt) {
            v16h bfr = *(const v16h*)(pw1 + (((kt * DIM) + col) << 5) + (g << 4));
            c0 = __builtin_amdgcn_wmma_f32_16x16x32_f16(
                     false, afr[0][kt], false, bfr, (short)0, c0, false, false);
            c1 = __builtin_amdgcn_wmma_f32_16x16x32_f16(
                     false, afr[1][kt], false, bfr, (short)0, c1, false, false);
        }
        #pragma unroll
        for (int r = 0; r < 8; ++r) {            // D layout: M = r + 8*g, N = col
            float v0 = c0[r] > 0.0f ? c0[r] : 0.0f;
            float v1 = c1[r] > 0.0f ? c1[r] : 0.0f;
            s_hid[w][r + 8 * g][col]      = (_Float16)v0;
            s_hid[w][16 + r + 8 * g][col] = (_Float16)v1;
        }
    }
    __syncthreads();

    // A fragments for GEMM2 from hidden [32x128]
    v16h afr2[2][4];
    #pragma unroll
    for (int mt = 0; mt < 2; ++mt) {
        #pragma unroll
        for (int kt = 0; kt < 4; ++kt) {
            const int kb = kt * 32;
            v8h lo = *(const v8h*)&s_hid[w][mt * 16 + nl][kb + g * 16];
            v8h hi = *(const v8h*)&s_hid[w][mt * 16 + nl][kb + g * 16 + 8];
            v16h a;
            #pragma unroll
            for (int i = 0; i < 8; ++i) { a[i] = lo[i]; a[8 + i] = hi[i]; }
            afr2[mt][kt] = a;
        }
    }

    // GEMM2: m[32x128] = hidden @ W2 + b2, then atomic segment-max scatter
    for (int nt = 0; nt < 8; ++nt) {
        const int col = nt * 16 + nl;
        const float bv = b2[col];
        v8f c0, c1;
        #pragma unroll
        for (int r = 0; r < 8; ++r) { c0[r] = bv; c1[r] = bv; }
        #pragma unroll
        for (int kt = 0; kt < 4; ++kt) {
            v16h bfr = *(const v16h*)(pw2 + (((kt * DIM) + col) << 5) + (g << 4));
            c0 = __builtin_amdgcn_wmma_f32_16x16x32_f16(
                     false, afr2[0][kt], false, bfr, (short)0, c0, false, false);
            c1 = __builtin_amdgcn_wmma_f32_16x16x32_f16(
                     false, afr2[1][kt], false, bfr, (short)0, c1, false, false);
        }
        #pragma unroll
        for (int r = 0; r < 8; ++r) {
            const int n0 = s_dst[w][r + 8 * g];
            const int n1 = s_dst[w][16 + r + 8 * g];
            atomicMax(&keys[(size_t)n0 * DIM + col], fkey(c0[r]));
            atomicMax(&keys[(size_t)n1 * DIM + col], fkey(c1[r]));
        }
    }
}

extern "C" void kernel_launch(void* const* d_in, const int* in_sizes, int n_in,
                              void* d_out, int out_size, void* d_ws, size_t ws_size,
                              hipStream_t stream) {
    const float*     z    = (const float*)d_in[0];
    const long long* eidx = (const long long*)d_in[1];  // int64 per reference
    const float*     W1   = (const float*)d_in[2];
    const float*     b1   = (const float*)d_in[3];
    const float*     W2   = (const float*)d_in[4];
    const float*     b2   = (const float*)d_in[5];

    // workspace carve-out (256B aligned)
    char*  ws  = (char*)d_ws;
    size_t off = 0;
    auto carve = [&](size_t bytes) -> void* {
        off = (off + 255) & ~(size_t)255;
        void* p = ws + off;
        off += bytes;
        return p;
    };
    unsigned* keys = (unsigned*)carve((size_t)N_NODES * DIM * sizeof(unsigned));
    float*    hA   = (float*)carve((size_t)N_NODES * DIM * sizeof(float));
    float*    hB   = (float*)carve((size_t)N_NODES * DIM * sizeof(float));
    _Float16* pw1  = (_Float16*)carve((size_t)NLAYER * 2 * DIM * DIM * sizeof(_Float16));
    _Float16* pw2  = (_Float16*)carve((size_t)NLAYER * DIM * DIM * sizeof(_Float16));
    (void)ws_size; (void)in_sizes; (void)n_in; (void)out_size;

    pack_weights_kernel<<<512, 256, 0, stream>>>(W1, W2, pw1, pw2);

    const float* hin = z;
    float* bufs[2] = {hA, hB};
    for (int l = 0; l < NLAYER; ++l) {
        init_keys_kernel<<<2048, 256, 0, stream>>>(keys);
        edge_mlp_kernel<<<NEDGE / 64, 64, 0, stream>>>(
            hin, eidx,
            pw1 + (size_t)l * 8 * DIM * 32,   // = l * 2*DIM*DIM halfs
            b1 + l * DIM,
            pw2 + (size_t)l * 4 * DIM * 32,   // = l * DIM*DIM halfs
            b2 + l * DIM,
            keys);
        float* hout = (l == NLAYER - 1) ? (float*)d_out : bufs[l & 1];
        finalize_kernel<<<2048, 256, 0, stream>>>(keys, hout,
                                                  (l < NLAYER - 1) ? 1 : 0);
        hin = hout;
    }
}
